// MutiHeadAttention_65618510348743
// MI455X (gfx1250) — matrix operand order repaired and checked by reading, not compiled, not run on hardware
//
#include <hip/hip_runtime.h>
#include <hip/hip_bf16.h>

typedef __bf16 bf16_t;
typedef __attribute__((ext_vector_type(16))) __bf16 v16b;
typedef __attribute__((ext_vector_type(8)))  __bf16 v8b;
typedef __attribute__((ext_vector_type(4)))  __bf16 v4b;
typedef __attribute__((ext_vector_type(8)))  float  v8f;
typedef __attribute__((ext_vector_type(4)))  float  v4f;
typedef __attribute__((ext_vector_type(4)))  unsigned u32x4;
typedef __attribute__((ext_vector_type(8)))  unsigned u32x8;

// Problem constants
#define Bc   4
#define Sc   8192
#define HIDc 1024
#define Hc   16
#define DHc  64
#define Mtot (Bc * Sc)   // 32768

// ---------------------------------------------------------------------------
// CDNA5 helpers: TDM descriptor load, async global->LDS copy, waits, swizzle
// ---------------------------------------------------------------------------
__device__ __forceinline__ unsigned lds_off(const void* p) {
    // generic pointers into LDS carry the byte offset in the low 32 bits
    return (unsigned)(uintptr_t)p;
}

// 2D tile load via Tensor Data Mover (D# groups per CDNA5 ISA 8.3/8.4).
// Copies tile_d1 rows of tile_d0 elements (2-byte) from global (row stride
// stride_elems) into LDS linearly at lds_byte_off. Tracked by TENSORcnt.
__device__ __forceinline__ void tdm_load_2d_bf16(unsigned lds_byte_off,
                                                 const void* gaddr,
                                                 unsigned tensor_d0,
                                                 unsigned tensor_d1,
                                                 unsigned stride_elems,
                                                 unsigned tile_d0,
                                                 unsigned tile_d1)
{
    unsigned long long ga = (unsigned long long)(uintptr_t)gaddr;
    u32x4 g0;
    g0[0] = 1u;                                   // count=1, user D#, no gather
    g0[1] = lds_byte_off;                         // lds_addr
    g0[2] = (unsigned)ga;                         // global_addr[31:0]
    g0[3] = (unsigned)((ga >> 32) & 0x1FFFFFFu)   // global_addr[56:32]
          | (2u << 30);                           // type=2 ("image")
    u32x8 g1;
    g1[0] = 1u << 16;                             // data_size=1 -> 2 bytes
    g1[1] = (tensor_d0 & 0xFFFFu) << 16;          // tensor_dim0[15:0]
    g1[2] = (tensor_d0 >> 16) | ((tensor_d1 & 0xFFFFu) << 16);
    g1[3] = (tensor_d1 >> 16) | (tile_d0 << 16);  // tile_dim0
    g1[4] = tile_d1 & 0xFFFFu;                    // tile_dim1 (tile_dim2 = 0)
    g1[5] = stride_elems;                         // tensor_dim0_stride[31:0]
    g1[6] = 0;
    g1[7] = 0;
    asm volatile("tensor_load_to_lds %0, %1" :: "s"(g0), "s"(g1) : "memory");
}

// Per-lane async 16-byte global->LDS copy (ASYNCcnt tracked)
__device__ __forceinline__ void async_copy_b128(void* lds_p, const void* gp) {
    asm volatile("global_load_async_to_lds_b128 %0, %1, off"
                 :: "v"(lds_off(lds_p)),
                    "v"((unsigned long long)(uintptr_t)gp)
                 : "memory");
}
__device__ __forceinline__ void wait_asynccnt0() {
    asm volatile("s_wait_asynccnt 0x0" ::: "memory");
}

// Butterfly XOR lane exchange via ds_swizzle group-of-32 mode:
// offset = xor_mask<<10 | or_mask<<5 | and_mask ; and_mask=0x1f, or=0.
template <int XORMASK>
__device__ __forceinline__ float swz_xor(float x) {
    return __int_as_float(
        __builtin_amdgcn_ds_swizzle(__float_as_int(x), (XORMASK << 10) | 0x1f));
}

// ---------------------------------------------------------------------------
// fp32 -> bf16 conversion (vectorized, grid-stride)
// ---------------------------------------------------------------------------
__global__ __launch_bounds__(256) void cvt_f32_to_bf16(const float* __restrict__ in,
                                                       bf16_t* __restrict__ out,
                                                       int n4) {
    int i = blockIdx.x * blockDim.x + threadIdx.x;
    int stride = gridDim.x * blockDim.x;
    for (; i < n4; i += stride) {
        v4f x = *(const v4f*)(in + (size_t)i * 4);
        v4b y;
        y[0] = (bf16_t)x[0];
        y[1] = (bf16_t)x[1];
        y[2] = (bf16_t)x[2];
        y[3] = (bf16_t)x[3];
        *(v4b*)(out + (size_t)i * 4) = y;
    }
}

// ---------------------------------------------------------------------------
// Tiled bf16 WMMA GEMM:  C[M,N] = A[M,K] (bf16) @ W[K,N] (bf16) + bias (f32)
// Block tile 128x128, K-step 32. 256 threads = 8 waves, wave tile 32x64.
// A tile staged by the TDM (double-buffered); B tile staged manually with a
// transpose so B-fragments are contiguous per the ISA 32x16 bf16 layout.
// ---------------------------------------------------------------------------
#define TM 128
#define TN 128
#define TK 32

template <int STORE_F32>
__global__ __launch_bounds__(256) void gemm_bf16_wmma(
    const bf16_t* __restrict__ A, const bf16_t* __restrict__ W,
    const float* __restrict__ bias,
    float* __restrict__ Cf, bf16_t* __restrict__ Cb,
    int M, int N, int K)
{
    __shared__ bf16_t As[2][TM * TK];   // row-major, stride TK (TDM-filled)
    __shared__ bf16_t BsT[2][TN * TK];  // n-major (transposed), stride TK

    const int tid  = threadIdx.x;
    const int lane = tid & 31;
    const int wid  = tid >> 5;       // 0..7
    const int wm   = wid & 3;        // row block of 32 within tile
    const int wn   = wid >> 2;       // col block of 64 within tile
    const int bm   = blockIdx.y * TM;
    const int bn   = blockIdx.x * TN;

    v8f acc[2][4];
#pragma unroll
    for (int i = 0; i < 2; i++)
#pragma unroll
        for (int j = 0; j < 4; j++)
#pragma unroll
            for (int r = 0; r < 8; r++) acc[i][j][r] = 0.0f;

    // B-tile global-load work split
    const int bkrow = tid >> 3;        // 0..31
    const int bncol = (tid & 7) * 16;  // 0..112

    const int ma  = lane & 15;
    const int kbA = (lane < 16) ? 0 : 8;   // A frag K-base (ISA 16x32 bf16 layout)
    const int kbB = (lane < 16) ? 0 : 8;   // B frag K-base (same table order as A)

    // ---- prologue: stage tile 0 into buffer 0 ----
    if (wid == 0)
        tdm_load_2d_bf16(lds_off(&As[0][0]), A + (size_t)bm * K,
                         (unsigned)K, TM, (unsigned)K, TK, TM);
    {
        const bf16_t* bg = W + (size_t)bkrow * N + bn + bncol;
        v8b b0 = *(const v8b*)(bg);
        v8b b1 = *(const v8b*)(bg + 8);
#pragma unroll
        for (int i = 0; i < 8; i++) BsT[0][(bncol + i) * TK + bkrow]     = b0[i];
#pragma unroll
        for (int i = 0; i < 8; i++) BsT[0][(bncol + 8 + i) * TK + bkrow] = b1[i];
    }
    if (wid == 0) __builtin_amdgcn_s_wait_tensorcnt(0);
    __syncthreads();

    int buf = 0;
    for (int k0 = 0; k0 < K; k0 += TK) {
        const bool has_next = (k0 + TK) < K;

        // prefetch the W tile two steps ahead into GL2 (global_prefetch_b8)
        if (k0 + 2 * TK < K)
            __builtin_prefetch(W + (size_t)(k0 + 2 * TK + bkrow) * N + bn + bncol, 0, 0);

        // fragments from current buffer
        v16b af[2];
#pragma unroll
        for (int i = 0; i < 2; i++) {
            const bf16_t* p = &As[buf][(wm * 32 + i * 16 + ma) * TK + kbA];
            v8b lo = *(const v8b*)(p);
            v8b hi = *(const v8b*)(p + 16);
#pragma unroll
            for (int e = 0; e < 8; e++) { af[i][e] = lo[e]; af[i][8 + e] = hi[e]; }
        }
        v16b bfr[4];
#pragma unroll
        for (int j = 0; j < 4; j++) {
            const bf16_t* p = &BsT[buf][(wn * 64 + j * 16 + ma) * TK + kbB];
            v8b lo = *(const v8b*)(p);
            v8b hi = *(const v8b*)(p + 16);
#pragma unroll
            for (int e = 0; e < 8; e++) { bfr[j][e] = lo[e]; bfr[j][8 + e] = hi[e]; }
        }

        // stage next tile into the other buffer while we compute
        if (has_next) {
            if (wid == 0)
                tdm_load_2d_bf16(lds_off(&As[buf ^ 1][0]),
                                 A + (size_t)bm * K + (k0 + TK),
                                 (unsigned)K, TM, (unsigned)K, TK, TM);
            const bf16_t* bg = W + (size_t)(k0 + TK + bkrow) * N + bn + bncol;
            v8b b0 = *(const v8b*)(bg);
            v8b b1 = *(const v8b*)(bg + 8);
#pragma unroll
            for (int i = 0; i < 8; i++) BsT[buf ^ 1][(bncol + i) * TK + bkrow]     = b0[i];
#pragma unroll
            for (int i = 0; i < 8; i++) BsT[buf ^ 1][(bncol + 8 + i) * TK + bkrow] = b1[i];
        }

#pragma unroll
        for (int i = 0; i < 2; i++)
#pragma unroll
            for (int j = 0; j < 4; j++)
                acc[i][j] = __builtin_amdgcn_wmma_f32_16x16x32_bf16(
                    false, af[i], false, bfr[j], (short)0, acc[i][j], false, false);

        if (has_next && wid == 0) __builtin_amdgcn_s_wait_tensorcnt(0);
        __syncthreads();
        buf ^= 1;
    }

    // epilogue: bias + store (D layout: VGPR r -> M=r+8g, N=lane&15)
    const int g  = lane >> 4;
    const int cn = lane & 15;
#pragma unroll
    for (int j = 0; j < 4; j++) {
        const int   n  = bn + wn * 64 + j * 16 + cn;
        const float bv = bias[n];
#pragma unroll
        for (int i = 0; i < 2; i++) {
            const int mbase = bm + wm * 32 + i * 16 + g * 8;
#pragma unroll
            for (int r = 0; r < 8; r++) {
                const float  val = acc[i][j][r] + bv;
                const size_t idx = (size_t)(mbase + r) * N + n;
                if (STORE_F32) Cf[idx] = val;
                else           Cb[idx] = (bf16_t)val;
            }
        }
    }
}

// ---------------------------------------------------------------------------
// Per-position cross-head attention. One wave per position (4 waves / block).
// scores = q(16x64) @ k^T(64x16) / 8 ; softmax over 16 ; out = probs @ v.
// q/k rows staged with async global->LDS copies (ASYNCcnt); softmax butterfly
// uses ds_swizzle XOR mode. Output stored bf16 directly in the reference's
// scrambled layout: row = b*8192 + h*512 + s/16, col = (s%16)*64 + d
// ---------------------------------------------------------------------------
__global__ __launch_bounds__(128) void attn_perpos(
    const bf16_t* __restrict__ Q, const bf16_t* __restrict__ Kq,
    const bf16_t* __restrict__ V, bf16_t* __restrict__ O)
{
    __shared__ bf16_t qs[4][1024];
    __shared__ bf16_t ks[4][1024];
    __shared__ bf16_t vts[4][1024];  // v transposed: vT[d*16 + h]
    __shared__ float  ps[4][256];    // probs 16x16

    const int lane = threadIdx.x & 31;
    const int w    = threadIdx.x >> 5;
    const int pos  = blockIdx.x * 4 + w;   // 0..32767
    const int b    = pos >> 13;            // / 8192
    const int s    = pos & 8191;

    const bf16_t* qg = Q + (size_t)pos * 1024;
    const bf16_t* kg = Kq + (size_t)pos * 1024;
    const bf16_t* vg = V + (size_t)pos * 1024;

#pragma unroll
    for (int i = 0; i < 4; i++) {
        const int e = i * 256 + lane * 8;
        async_copy_b128(&qs[w][e], qg + e);   // straight copies: async path
        async_copy_b128(&ks[w][e], kg + e);
        v8b vv = *(const v8b*)(vg + e);       // v needs transpose: manual
        const int h = e >> 6, d0 = e & 63;
#pragma unroll
        for (int j = 0; j < 8; j++) vts[w][(d0 + j) * 16 + h] = vv[j];
    }
    wait_asynccnt0();
    __syncthreads();

    const int m   = lane & 15;
    const int g   = lane >> 4;
    const int kbA = (lane < 16) ? 0 : 8;
    const int kbB = (lane < 16) ? 0 : 8;

    // scores = q @ k^T  (K = 64 -> two K=32 WMMAs)
    v8f sc;
#pragma unroll
    for (int r = 0; r < 8; r++) sc[r] = 0.0f;
#pragma unroll
    for (int c = 0; c < 2; c++) {
        v16b aq, bk;
        const bf16_t* pa = &qs[w][m * 64 + c * 32 + kbA];
        v8b lo = *(const v8b*)(pa);
        v8b hi = *(const v8b*)(pa + 16);
#pragma unroll
        for (int e = 0; e < 8; e++) { aq[e] = lo[e]; aq[8 + e] = hi[e]; }
        const bf16_t* pb = &ks[w][m * 64 + c * 32 + kbB];
        v8b l2 = *(const v8b*)(pb);
        v8b h2 = *(const v8b*)(pb + 16);
#pragma unroll
        for (int e = 0; e < 8; e++) { bk[e] = l2[e]; bk[8 + e] = h2[e]; }
        sc = __builtin_amdgcn_wmma_f32_16x16x32_bf16(false, aq, false, bk,
                                                     (short)0, sc, false, false);
    }

    // softmax over the 16 columns (ds_swizzle XOR butterfly within 16 lanes)
#pragma unroll
    for (int r = 0; r < 8; r++) {
        float x = sc[r] * 0.125f;  // 1/sqrt(64)
        float mx = x;
        mx = fmaxf(mx, swz_xor<8>(mx));
        mx = fmaxf(mx, swz_xor<4>(mx));
        mx = fmaxf(mx, swz_xor<2>(mx));
        mx = fmaxf(mx, swz_xor<1>(mx));
        float ex = __expf(x - mx);
        float sum = ex;
        sum += swz_xor<8>(sum);
        sum += swz_xor<4>(sum);
        sum += swz_xor<2>(sum);
        sum += swz_xor<1>(sum);
        ps[w][(r + 8 * g) * 16 + m] = ex / sum;
    }
    __syncthreads();

    // A = probs (16x16), K-padded to 32 with zeros
    v16b ap;
    {
        const float* pp = &ps[w][m * 16 + kbA];
#pragma unroll
        for (int e = 0; e < 8; e++) { ap[e] = (bf16_t)pp[e]; ap[8 + e] = (bf16_t)0.0f; }
    }

    // out = probs @ v : 4 chunks of 16 columns of DH
#pragma unroll
    for (int c = 0; c < 4; c++) {
        v16b bv;
#pragma unroll
        for (int e = 0; e < 16; e++) bv[e] = (bf16_t)0.0f;
        {  // K = 0..15 valid (heads, k = kbA + e); K >= 16 is padding
            const bf16_t* pv = &vts[w][(c * 16 + m) * 16 + kbA];
            v8b lo = *(const v8b*)(pv);
#pragma unroll
            for (int e = 0; e < 8; e++) { bv[e] = lo[e]; }
        }
        v8f o;
#pragma unroll
        for (int r = 0; r < 8; r++) o[r] = 0.0f;
        o = __builtin_amdgcn_wmma_f32_16x16x32_bf16(false, ap, false, bv,
                                                    (short)0, o, false, false);
        const int d = c * 16 + m;
#pragma unroll
        for (int r = 0; r < 8; r++) {
            const int    hm  = r + 8 * g;
            const size_t row = (size_t)(b * 8192 + hm * 512 + (s >> 4));
            O[row * 1024 + (size_t)((s & 15) * 64 + d)] = (bf16_t)o[r];
        }
    }
}

// ---------------------------------------------------------------------------
extern "C" void kernel_launch(void* const* d_in, const int* in_sizes, int n_in,
                              void* d_out, int out_size, void* d_ws, size_t ws_size,
                              hipStream_t stream)
{
    const float* hid = (const float*)d_in[0];
    const float* Wq  = (const float*)d_in[1];
    const float* bq  = (const float*)d_in[2];
    const float* Wk  = (const float*)d_in[3];
    const float* bk  = (const float*)d_in[4];
    const float* Wv  = (const float*)d_in[5];
    const float* bv  = (const float*)d_in[6];
    const float* Wo  = (const float*)d_in[7];
    const float* bo  = (const float*)d_in[8];
    float* out = (float*)d_out;

    const int M = Mtot, N = HIDc, K = HIDc;

    char* wsp = (char*)d_ws;
    bf16_t* hidb = (bf16_t*)wsp; wsp += (size_t)M * K * sizeof(bf16_t);
    bf16_t* Wqb  = (bf16_t*)wsp; wsp += (size_t)K * N * sizeof(bf16_t);
    bf16_t* Wkb  = (bf16_t*)wsp; wsp += (size_t)K * N * sizeof(bf16_t);
    bf16_t* Wvb  = (bf16_t*)wsp; wsp += (size_t)K * N * sizeof(bf16_t);
    bf16_t* Wob  = (bf16_t*)wsp; wsp += (size_t)K * N * sizeof(bf16_t);
    bf16_t* qb   = (bf16_t*)wsp; wsp += (size_t)M * N * sizeof(bf16_t);
    bf16_t* kb2  = (bf16_t*)wsp; wsp += (size_t)M * N * sizeof(bf16_t);
    bf16_t* vb2  = (bf16_t*)wsp; wsp += (size_t)M * N * sizeof(bf16_t);
    bf16_t* aob  = (bf16_t*)wsp; wsp += (size_t)M * N * sizeof(bf16_t);

    // fp32 -> bf16 conversions
    cvt_f32_to_bf16<<<2048, 256, 0, stream>>>(hid, hidb, (M * K) / 4);
    cvt_f32_to_bf16<<<512,  256, 0, stream>>>(Wq, Wqb, (K * N) / 4);
    cvt_f32_to_bf16<<<512,  256, 0, stream>>>(Wk, Wkb, (K * N) / 4);
    cvt_f32_to_bf16<<<512,  256, 0, stream>>>(Wv, Wvb, (K * N) / 4);
    cvt_f32_to_bf16<<<512,  256, 0, stream>>>(Wo, Wob, (K * N) / 4);

    dim3 gg(N / TN, M / TM);  // (8, 256)

    // Q/K/V projections (bf16 out)
    gemm_bf16_wmma<0><<<gg, 256, 0, stream>>>(hidb, Wqb, bq, nullptr, qb,  M, N, K);
    gemm_bf16_wmma<0><<<gg, 256, 0, stream>>>(hidb, Wkb, bk, nullptr, kb2, M, N, K);
    gemm_bf16_wmma<0><<<gg, 256, 0, stream>>>(hidb, Wvb, bv, nullptr, vb2, M, N, K);

    // per-position cross-head attention, output pre-scrambled
    attn_perpos<<<M / 4, 128, 0, stream>>>(qb, kb2, vb2, aob);

    // output projection (fp32 out)
    gemm_bf16_wmma<1><<<gg, 256, 0, stream>>>(aob, Wob, bo, out, nullptr, M, N, K);
}
